// AutoformerEncoderLayer_86938728006128
// MI455X (gfx1250) — matrix operand order repaired and checked by reading, not compiled, not run on hardware
//
#include <hip/hip_runtime.h>
#include <hip/hip_fp16.h>
#include <hip/hip_bf16.h>
#include <cstdint>

typedef __attribute__((ext_vector_type(16))) _Float16 v16h;
typedef __attribute__((ext_vector_type(8)))  _Float16 v8h;
typedef __attribute__((ext_vector_type(8)))  float    v8f;

namespace {

constexpr int kB   = 8;
constexpr int kL   = 2048;
constexpr int kDm  = 512;
constexpr int kH   = 8;
constexpr int kD   = 64;
constexpr int kM   = kB * kL;      // 16384 token rows
constexpr float kLnEps = 1e-5f;

constexpr int MODE_HEADS  = 0;   // +bias, f16 store scattered to (B,H,L,D)
constexpr int MODE_F32    = 1;   // +bias, f32 store row-major
constexpr int MODE_ADDRES = 2;   // +bias +residual, f32 store row-major
constexpr int MODE_GELU   = 3;   // +bias, exact GELU, f16 store row-major

#define WMMA_F16F32(a, b, c) \
  __builtin_amdgcn_wmma_f32_16x16x32_f16(false, (a), false, (b), (short)0, (c), false, false)

// ---- fragment loaders (16-bit WMMA operands, contiguous k = 16*half + s in both A and B) ----
// A 16x32 f16: lane = 16*half + m ; elements 0..15 = K[k0+16h .. +15]
__device__ inline v16h load_a_f16(const _Float16* __restrict__ a, int ld, int m0, int k0) {
  const int lane = threadIdx.x & 31;
  const _Float16* p = a + (size_t)(m0 + (lane & 15)) * ld + k0 + ((lane >> 4) << 4);
  v8h lo = *(const v8h*)p;
  v8h hi = *(const v8h*)(p + 8);
  v16h r;
#pragma unroll
  for (int i = 0; i < 8; ++i) { r[i] = lo[i]; r[i + 8] = hi[i]; }
  return r;
}

// A with per-lane wrapped row index (circular correlation), L power of two
__device__ inline v16h load_a_f16_mod(const _Float16* __restrict__ a, int ld, int m0, int k0) {
  const int lane = threadIdx.x & 31;
  const int row = (m0 + (lane & 15)) & (kL - 1);
  const _Float16* p = a + (size_t)row * ld + k0 + ((lane >> 4) << 4);
  v8h lo = *(const v8h*)p;
  v8h hi = *(const v8h*)(p + 8);
  v16h r;
#pragma unroll
  for (int i = 0; i < 8; ++i) { r[i] = lo[i]; r[i + 8] = hi[i]; }
  return r;
}

// B 32x16 f16 from (N,K)-major storage: lane = 16*half + n ; 16 contiguous K at k0+16*half
__device__ inline v16h load_b_f16(const _Float16* __restrict__ bt, int ld, int n0, int k0) {
  const int lane = threadIdx.x & 31;
  const _Float16* p = bt + (size_t)(n0 + (lane & 15)) * ld + k0 + ((lane >> 4) << 4);
  return *(const v16h*)p;
}

__device__ inline float gelu_exact(float v) {
  return 0.5f * v * (1.0f + erff(v * 0.70710678118654752f));
}

// ---- f32 -> f16 bulk convert (amortizes A-operand conversion across GEMMs) ----
__global__ __launch_bounds__(256) void f32_to_f16_kernel(const float* __restrict__ in,
                                                         _Float16* __restrict__ out) {
  const int idx = blockIdx.x * 256 + threadIdx.x;
  out[idx] = (_Float16)in[idx];
}

// ---- weight transpose + f16 convert: W (K,N) row-major -> Wt (N,K) row-major ----
__global__ __launch_bounds__(256) void transpose_f16_kernel(const float* __restrict__ W,
                                                            _Float16* __restrict__ Wt,
                                                            int K, int N) {
  const int idx = blockIdx.x * 256 + threadIdx.x;   // idx = n*K + k
  const int n = idx / K;
  const int k = idx - n * K;
  Wt[idx] = (_Float16)W[(size_t)k * N + n];
}

// ---- generic WMMA GEMM: C(M,N) = A(M,K) @ Bt(N,K)^T + bias, epilogue by MODE ----
// 8 waves; wave tile 32x64 (8 WMMA chains); block tile 256x64.
// B tile staged workgroup-wide into LDS via async DMA (ASYNCcnt), double buffered.
template <int MODE>
__global__ __launch_bounds__(256) void gemm_wmma_kernel(const _Float16* __restrict__ A,
                                                        const _Float16* __restrict__ Bt,
                                                        const float* __restrict__ bias,
                                                        const float* __restrict__ res,
                                                        void* __restrict__ outp,
                                                        int N, int K) {
  constexpr int BROW = 64;   // staged n-rows
  constexpr int BSTR = 40;   // halfs per LDS row (80B pitch: stride 20 banks, conflict-free)
  __shared__ _Float16 bsh[2][BROW * BSTR];

  const int wave = threadIdx.x >> 5;
  const int lane = threadIdx.x & 31;
  const int m0 = blockIdx.y * 256 + wave * 32;
  const int n0 = blockIdx.x * 64;

  // staging map: thread -> (row 0..63, 16B segment 0..3); one async b128 per thread per tile
  const int srow = threadIdx.x >> 2;
  const int sseg = threadIdx.x & 3;
  const _Float16* gsrc = Bt + (size_t)(n0 + srow) * K + sseg * 8;
  const uint32_t lds0 = (uint32_t)(uintptr_t)&bsh[0][srow * BSTR + sseg * 8];
  const uint32_t lds1 = (uint32_t)(uintptr_t)&bsh[1][srow * BSTR + sseg * 8];

  v8f acc[8] = {};

  // prologue: stage tile k0=0 into buffer 0
  asm volatile("global_load_async_to_lds_b128 %0, %1, off"
               :: "v"(lds0), "v"(gsrc) : "memory");

  int cur = 0;
  for (int k0 = 0; k0 < K; k0 += 32) {
    const bool more = (k0 + 32) < K;
    if (more) {
      const uint32_t ldsn = cur ? lds0 : lds1;
      const _Float16* gn = gsrc + k0 + 32;
      asm volatile("global_load_async_to_lds_b128 %0, %1, off"
                   :: "v"(ldsn), "v"(gn) : "memory");
      asm volatile("s_wait_asynccnt 0x1" ::: "memory");   // oldest tile landed
    } else {
      asm volatile("s_wait_asynccnt 0x0" ::: "memory");
    }
    __syncthreads();

    v16h a0 = load_a_f16(A, K, m0, k0);
    v16h a1 = load_a_f16(A, K, m0 + 16, k0);
#pragma unroll
    for (int t = 0; t < 4; ++t) {
      const _Float16* bp = &bsh[cur][(t * 16 + (lane & 15)) * BSTR + ((lane >> 4) << 4)];
      v16h b = *(const v16h*)bp;      // 2x ds_load_b128
      acc[t]     = WMMA_F16F32(a0, b, acc[t]);
      acc[t + 4] = WMMA_F16F32(a1, b, acc[t + 4]);
    }
    __syncthreads();                  // tile fully consumed before DMA may overwrite
    cur ^= 1;
  }

  // C layout: element (m = v + 8*half, n = lane&15); acc[tt]: ntile = tt&3, mtile = tt>>2
  const int half = lane >> 4;
  const int j = lane & 15;
#pragma unroll
  for (int tt = 0; tt < 8; ++tt) {
    const int n = n0 + (tt & 3) * 16 + j;
    const int mbase = m0 + (tt >> 2) * 16 + half * 8;
    const float bval = bias[n];
#pragma unroll
    for (int v = 0; v < 8; ++v) {
      const int m = mbase + v;
      const float val = acc[tt][v] + bval;
      if (MODE == MODE_HEADS) {
        const int bb = m >> 11, l = m & (kL - 1), h = n >> 6, d = n & 63;
        ((_Float16*)outp)[((size_t)((bb * kH + h) * kL + l)) * kD + d] = (_Float16)val;
      } else if (MODE == MODE_F32) {
        ((float*)outp)[(size_t)m * N + n] = val;
      } else if (MODE == MODE_ADDRES) {
        ((float*)outp)[(size_t)m * N + n] = val + res[(size_t)m * N + n];
      } else {  // MODE_GELU
        ((_Float16*)outp)[(size_t)m * N + n] = (_Float16)gelu_exact(val);
      }
    }
  }
}

// ---- circular autocorrelation mean via banded Q@K^T + diagonal reduction ----
// corr[bh, tau] = (1/D) * sum_s Q[(s+tau)%L] . K[s]
// wave handles one 16-wide tau block; accumulates two r-block tile chains over all s;
// the diagonal index (i-j)&15 is s-independent, so the scatter happens once at the end.
__global__ __launch_bounds__(128) void corr_wmma_kernel(const _Float16* __restrict__ Qh,
                                                        const _Float16* __restrict__ Kh,
                                                        float* __restrict__ corr) {
  const int wave = threadIdx.x >> 5;
  const int lane = threadIdx.x & 31;
  const int bh   = blockIdx.x >> 5;                 // 32 blocks per (b,h)
  const int taub = (blockIdx.x & 31) * 64;          // 4 waves x 16 tau each
  const int tau0 = taub + wave * 16;

  const _Float16* Qb = Qh + (size_t)bh * kL * kD;
  const _Float16* Kb = Kh + (size_t)bh * kL * kD;

  v8f acc1 = {};   // rows r = s0+tau0 .. +15   -> diagonals i>=j
  v8f acc2 = {};   // rows r = s0+tau0+16 .. +31 -> diagonals i<j
  for (int s0 = 0; s0 < kL; s0 += 16) {
    const int r0 = s0 + tau0;
    v16h b0  = load_b_f16(Kb, kD, s0, 0);
    v16h b1  = load_b_f16(Kb, kD, s0, 32);
    v16h a10 = load_a_f16_mod(Qb, kD, r0, 0);
    v16h a11 = load_a_f16_mod(Qb, kD, r0, 32);
    v16h a20 = load_a_f16_mod(Qb, kD, r0 + 16, 0);
    v16h a21 = load_a_f16_mod(Qb, kD, r0 + 16, 32);
    acc1 = WMMA_F16F32(a10, b0, acc1);
    acc1 = WMMA_F16F32(a11, b1, acc1);
    acc2 = WMMA_F16F32(a20, b0, acc2);
    acc2 = WMMA_F16F32(a21, b1, acc2);
  }

  __shared__ float lacc[4][16];
  if (threadIdx.x < 64) lacc[threadIdx.x >> 4][threadIdx.x & 15] = 0.f;
  __syncthreads();

  const int half = lane >> 4;
  const int j = lane & 15;
#pragma unroll
  for (int v = 0; v < 8; ++v) {
    const int i = half * 8 + v;
    const int d = (i - j) & 15;                     // tau - tau0
    const float val = (i >= j) ? acc1[v] : acc2[v];
    atomicAdd(&lacc[wave][d], val);
  }
  __syncthreads();
  if (threadIdx.x < 64) {
    const int w = threadIdx.x >> 4, d = threadIdx.x & 15;
    corr[(size_t)bh * kL + taub + w * 16 + d] = lacc[w][d] * (1.f / (float)kD);
  }
}

// ---- top-3 + softmax per (b,h) ----
__global__ __launch_bounds__(256) void topk_kernel(const float* __restrict__ corr,
                                                   float* __restrict__ wts,
                                                   int* __restrict__ dly) {
  const int bh = blockIdx.x;
  const int tid = threadIdx.x;
  const float* p = corr + (size_t)bh * kL;
  float v0 = -1e30f, v1 = -1e30f, v2 = -1e30f;
  int   i0 = 0, i1 = 0, i2 = 0;
  for (int l = tid; l < kL; l += 256) {
    const float v = p[l];
    if (v > v0)      { v2 = v1; i2 = i1; v1 = v0; i1 = i0; v0 = v; i0 = l; }
    else if (v > v1) { v2 = v1; i2 = i1; v1 = v;  i1 = l; }
    else if (v > v2) { v2 = v;  i2 = l; }
  }
  __shared__ float sv[768];
  __shared__ int   si[768];
  sv[tid] = v0; sv[tid + 256] = v1; sv[tid + 512] = v2;
  si[tid] = i0; si[tid + 256] = i1; si[tid + 512] = i2;
  __syncthreads();
  if (tid == 0) {
    float b0 = -1e30f, b1 = -1e30f, b2 = -1e30f;
    int   j0 = 0, j1 = 0, j2 = 0;
    for (int i = 0; i < 768; ++i) {
      const float v = sv[i]; const int ix = si[i];
      if (v > b0)      { b2 = b1; j2 = j1; b1 = b0; j1 = j0; b0 = v; j0 = ix; }
      else if (v > b1) { b2 = b1; j2 = j1; b1 = v;  j1 = ix; }
      else if (v > b2) { b2 = v;  j2 = ix; }
    }
    const float e0 = 1.f, e1 = expf(b1 - b0), e2 = expf(b2 - b0);
    const float inv = 1.f / (e0 + e1 + e2);
    wts[bh * 3 + 0] = e0 * inv; wts[bh * 3 + 1] = e1 * inv; wts[bh * 3 + 2] = e2 * inv;
    dly[bh * 3 + 0] = j0;       dly[bh * 3 + 1] = j1;       dly[bh * 3 + 2] = j2;
  }
}

// ---- delay-roll weighted combine: ctx[b,l,:] = sum_k w_k * V[b,(l-delay_k)%L,:] ----
__global__ __launch_bounds__(256) void roll_combine_kernel(const float* __restrict__ V32,
                                                           const float* __restrict__ wts,
                                                           const int* __restrict__ dly,
                                                           _Float16* __restrict__ ctx) {
  const int idx = blockIdx.x * 256 + threadIdx.x;   // over B*L*Dm
  const int n = idx & (kDm - 1);
  const int l = (idx >> 9) & (kL - 1);
  const int b = idx >> 20;
  const int base = (b * kH + (n >> 6)) * 3;
  float acc = 0.f;
#pragma unroll
  for (int k = 0; k < 3; ++k) {
    const int ls = (l - dly[base + k]) & (kL - 1);
    acc += wts[base + k] * V32[((size_t)(b * kL + ls)) * kDm + n];
  }
  ctx[idx] = (_Float16)acc;
}

// ---- LayerNorm over last dim (512); optional fused row-mean of the output ----
__global__ __launch_bounds__(256) void layernorm_kernel(const float* __restrict__ in,
                                                        const float* __restrict__ g,
                                                        const float* __restrict__ be,
                                                        float* __restrict__ out,
                                                        float* __restrict__ rowmean) {
  __shared__ float red[256];
  const int row = blockIdx.x;
  const int tid = threadIdx.x;
  const float* p = in + (size_t)row * kDm;
  const float x0 = p[tid], x1 = p[tid + 256];

  red[tid] = x0 + x1;
  __syncthreads();
  for (int o = 128; o > 0; o >>= 1) { if (tid < o) red[tid] += red[tid + o]; __syncthreads(); }
  const float mu = red[0] * (1.f / kDm);
  __syncthreads();

  const float d0 = x0 - mu, d1 = x1 - mu;
  red[tid] = d0 * d0 + d1 * d1;
  __syncthreads();
  for (int o = 128; o > 0; o >>= 1) { if (tid < o) red[tid] += red[tid + o]; __syncthreads(); }
  const float rinv = rsqrtf(red[0] * (1.f / kDm) + kLnEps);

  const float o0 = d0 * rinv * g[tid]       + be[tid];
  const float o1 = d1 * rinv * g[tid + 256] + be[tid + 256];
  float* q = out + (size_t)row * kDm;
  q[tid] = o0; q[tid + 256] = o1;

  if (rowmean != nullptr) {
    __syncthreads();
    red[tid] = o0 + o1;
    __syncthreads();
    for (int o = 128; o > 0; o >>= 1) { if (tid < o) red[tid] += red[tid + o]; __syncthreads(); }
    if (tid == 0) rowmean[row] = red[0] * (1.f / kDm);
  }
}

// ---- moving average (k=25, zero pad, count_include_pad) over L per batch ----
__global__ __launch_bounds__(256) void trend_kernel(const float* __restrict__ x2d,
                                                    float* __restrict__ trend) {
  const int idx = blockIdx.x * 256 + threadIdx.x;   // over B*L
  const int b = idx >> 11, l = idx & (kL - 1);
  const float* p = x2d + (size_t)b * kL;
  float s = 0.f;
#pragma unroll
  for (int j = -12; j <= 12; ++j) {
    const int t = l + j;
    if (t >= 0 && t < kL) s += p[t];
  }
  trend[idx] = s * (1.f / 25.f);
}

// ---- seasonal scale: xs = y * (1 + 0.1 * (x2d - trend)/(x2d + 1e-8)) ----
__global__ __launch_bounds__(256) void scale_kernel(const float* __restrict__ y,
                                                    const float* __restrict__ x2d,
                                                    const float* __restrict__ trend,
                                                    float* __restrict__ xs,
                                                    _Float16* __restrict__ xsh) {
  const int idx = blockIdx.x * 256 + threadIdx.x;   // over B*L*Dm
  const int row = idx >> 9;
  const float m = x2d[row];
  const float sc = (m - trend[row]) / (m + 1e-8f);
  const float v = y[idx] * (1.f + 0.1f * sc);
  xs[idx] = v;
  xsh[idx] = (_Float16)v;
}

}  // namespace

extern "C" void kernel_launch(void* const* d_in, const int* in_sizes, int n_in,
                              void* d_out, int out_size, void* d_ws, size_t ws_size,
                              hipStream_t stream) {
  (void)in_sizes; (void)n_in; (void)out_size; (void)ws_size;
  const float* x   = (const float*)d_in[0];
  const float* Wq  = (const float*)d_in[1];
  const float* bq  = (const float*)d_in[2];
  const float* Wk  = (const float*)d_in[3];
  const float* bk  = (const float*)d_in[4];
  const float* Wv  = (const float*)d_in[5];
  const float* bvp = (const float*)d_in[6];
  const float* Wo  = (const float*)d_in[7];
  const float* bo  = (const float*)d_in[8];
  const float* W1  = (const float*)d_in[9];
  const float* b1  = (const float*)d_in[10];
  const float* W2  = (const float*)d_in[11];
  const float* b2  = (const float*)d_in[12];
  const float* g1  = (const float*)d_in[13];
  const float* be1 = (const float*)d_in[14];
  const float* g2  = (const float*)d_in[15];
  const float* be2 = (const float*)d_in[16];

  char* ws = (char*)d_ws;
  size_t off = 0;
  auto carve = [&](size_t bytes) -> void* {
    void* p = ws + off;
    off += (bytes + 255) & ~(size_t)255;
    return p;
  };

  _Float16* wtq  = (_Float16*)carve((size_t)kDm * kDm * 2);
  _Float16* wtk  = (_Float16*)carve((size_t)kDm * kDm * 2);
  _Float16* wtv  = (_Float16*)carve((size_t)kDm * kDm * 2);
  _Float16* wto  = (_Float16*)carve((size_t)kDm * kDm * 2);
  _Float16* wt1  = (_Float16*)carve((size_t)kDm * 2048 * 2);
  _Float16* wt2  = (_Float16*)carve((size_t)2048 * kDm * 2);
  _Float16* xh   = (_Float16*)carve((size_t)kM * kDm * 2);
  _Float16* Qh   = (_Float16*)carve((size_t)kB * kH * kL * kD * 2);
  _Float16* Kh   = (_Float16*)carve((size_t)kB * kH * kL * kD * 2);
  float*    V32  = (float*)carve((size_t)kM * kDm * 4);
  float*    corr = (float*)carve((size_t)kB * kH * kL * 4);
  float*    wts  = (float*)carve((size_t)kB * kH * 3 * 4);
  int*      dly  = (int*)carve((size_t)kB * kH * 3 * 4);
  _Float16* ctx  = (_Float16*)carve((size_t)kM * kDm * 2);
  float*    pre1 = (float*)carve((size_t)kM * kDm * 4);   // residual+attn, LN1 in place
  float*    x2d  = (float*)carve((size_t)kM * 4);
  float*    trnd = (float*)carve((size_t)kM * 4);
  float*    xs   = (float*)carve((size_t)kM * kDm * 4);
  _Float16* xsh  = (_Float16*)carve((size_t)kM * kDm * 2);
  _Float16* hbuf = (_Float16*)carve((size_t)kM * 2048 * 2);
  float*    pre2 = (float*)carve((size_t)kM * kDm * 4);

  // weight transpose + f16 convert; x -> f16 once (reused by Q/K/V GEMMs)
  transpose_f16_kernel<<<(kDm * kDm) / 256, 256, 0, stream>>>(Wq, wtq, kDm, kDm);
  transpose_f16_kernel<<<(kDm * kDm) / 256, 256, 0, stream>>>(Wk, wtk, kDm, kDm);
  transpose_f16_kernel<<<(kDm * kDm) / 256, 256, 0, stream>>>(Wv, wtv, kDm, kDm);
  transpose_f16_kernel<<<(kDm * kDm) / 256, 256, 0, stream>>>(Wo, wto, kDm, kDm);
  transpose_f16_kernel<<<(kDm * 2048) / 256, 256, 0, stream>>>(W1, wt1, kDm, 2048);
  transpose_f16_kernel<<<(2048 * kDm) / 256, 256, 0, stream>>>(W2, wt2, 2048, kDm);
  f32_to_f16_kernel<<<(kM * kDm) / 256, 256, 0, stream>>>(x, xh);

  // Q/K projections -> f16 (B,H,L,D); V projection -> f32 (B*L, Dm)
  gemm_wmma_kernel<MODE_HEADS><<<dim3(8, 64), 256, 0, stream>>>(xh, wtq, bq, nullptr, Qh, kDm, kDm);
  gemm_wmma_kernel<MODE_HEADS><<<dim3(8, 64), 256, 0, stream>>>(xh, wtk, bk, nullptr, Kh, kDm, kDm);
  gemm_wmma_kernel<MODE_F32>  <<<dim3(8, 64), 256, 0, stream>>>(xh, wtv, bvp, nullptr, V32, kDm, kDm);

  // autocorrelation mean (replaces FFT): banded WMMA Q@K^T + diagonal reduce
  corr_wmma_kernel<<<kB * kH * (kL / 64), 128, 0, stream>>>(Qh, Kh, corr);

  // top-3 delays + softmax weights per (b,h)
  topk_kernel<<<kB * kH, 256, 0, stream>>>(corr, wts, dly);

  // weighted delay-roll of V -> f16 context
  roll_combine_kernel<<<(kM * kDm) / 256, 256, 0, stream>>>(V32, wts, dly, ctx);

  // output projection + residual, then LN1 (in place) with fused row-mean
  gemm_wmma_kernel<MODE_ADDRES><<<dim3(8, 64), 256, 0, stream>>>(ctx, wto, bo, x, pre1, kDm, kDm);
  layernorm_kernel<<<kM, 256, 0, stream>>>(pre1, g1, be1, pre1, x2d);

  // series decomposition scale (fused f32 + f16 outputs)
  trend_kernel<<<kM / 256, 256, 0, stream>>>(x2d, trnd);
  scale_kernel<<<(kM * kDm) / 256, 256, 0, stream>>>(pre1, x2d, trnd, xs, xsh);

  // FFN: gelu(xs @ W1 + b1) -> f16 ; @ W2 + b2 + xs ; LN2 -> d_out
  gemm_wmma_kernel<MODE_GELU>  <<<dim3(32, 64), 256, 0, stream>>>(xsh, wt1, b1, nullptr, hbuf, 2048, kDm);
  gemm_wmma_kernel<MODE_ADDRES><<<dim3(8, 64),  256, 0, stream>>>(hbuf, wt2, b2, xs, pre2, kDm, 2048);
  layernorm_kernel<<<kM, 256, 0, stream>>>(pre2, g2, be2, (float*)d_out, nullptr);
}
